// VectorQuantizer_28595892257049
// MI455X (gfx1250) — compile-verified, matrix operand order checked by reading, not run on hardware
//
#include <hip/hip_runtime.h>

#define BATCH   32768
#define EMB_DIM 256
#define N_EMB   8192

typedef __attribute__((ext_vector_type(16))) __bf16 v16bf;
typedef __attribute__((ext_vector_type(8)))  __bf16 v8bf;
typedef __attribute__((ext_vector_type(4)))  __bf16 v4bf;
typedef __attribute__((ext_vector_type(8)))  float  v8f;

// Load one 16x32 (MxK) bf16 WMMA operand fragment for this lane.
// ISA layout (CDNA5 ISA): lanes 0-15 carry K 0-7 (elems 0-7) and
// K 16-23 (elems 8-15); lanes 16-31 carry K 8-15 and K 24-31. Two b128 loads.
__device__ __forceinline__ v16bf load_frag(const __bf16* __restrict__ row,
                                           int kbase, int half) {
  v8bf lo = *(const v8bf*)(row + kbase + 8 * half);
  v8bf hi = *(const v8bf*)(row + kbase + 16 + 8 * half);
  return __builtin_shufflevector(lo, hi, 0, 1, 2, 3, 4, 5, 6, 7,
                                 8, 9, 10, 11, 12, 13, 14, 15);
}

// CDNA5 async copy global -> LDS, tracked by ASYNCcnt (no VGPR data path).
__device__ __forceinline__ void async_copy_b128(unsigned lds_off,
                                                const void* gaddr) {
  asm volatile("global_load_async_to_lds_b128 %0, %1, off"
               :: "v"(lds_off), "v"(gaddr) : "memory");
}
__device__ __forceinline__ void wait_async_le8() {
  asm volatile("s_wait_asynccnt 0x8" ::: "memory");
}
__device__ __forceinline__ void wait_async_0() {
  asm volatile("s_wait_asynccnt 0x0" ::: "memory");
}

// ---------------- prep: x -> bf16 ----------------
__global__ __launch_bounds__(256) void vq_prep_x(const float* __restrict__ x,
                                                 __bf16* __restrict__ xb) {
  int i = blockIdx.x * blockDim.x + threadIdx.x;   // one float4 per thread
  float4 v = ((const float4*)x)[i];
  v4bf o = {(__bf16)v.x, (__bf16)v.y, (__bf16)v.z, (__bf16)v.w};
  ((v4bf*)xb)[i] = o;
}

// ------- prep: codebook -> bf16, |e_k|^2 in f32, zero loss accumulator -------
__global__ __launch_bounds__(256) void vq_prep_e(const float* __restrict__ e,
                                                 __bf16* __restrict__ eb,
                                                 float* __restrict__ enorm,
                                                 float* __restrict__ acc) {
  if (blockIdx.x == 0 && threadIdx.x == 0) *acc = 0.0f;
  int wave = threadIdx.x >> 5, lane = threadIdx.x & 31;
  int row = blockIdx.x * 8 + wave;                 // one wave per codebook row
  const float4* s4 = (const float4*)(e + (size_t)row * EMB_DIM);
  float ssum = 0.0f;
#pragma unroll
  for (int i = 0; i < 2; ++i) {
    int c = lane + i * 32;                         // float4 index 0..63
    float4 v = s4[c];
    v4bf o = {(__bf16)v.x, (__bf16)v.y, (__bf16)v.z, (__bf16)v.w};
    ((v4bf*)eb)[(size_t)row * 64 + c] = o;
    ssum += v.x * v.x + v.y * v.y + v.z * v.z + v.w * v.w;
  }
#pragma unroll
  for (int m = 16; m >= 1; m >>= 1) ssum += __shfl_xor(ssum, m, 32);
  if (lane == 0) enorm[row] = ssum;
}

// -------- fused bf16-WMMA GEMM + row argmin over |e|^2 - 2 x.e --------
#define MTILE 64                  // rows per workgroup (4 waves x 16)
#define NTILE 32                  // codebook rows per LDS buffer
#define LDS_STRIDE 264            // bf16 per padded row: 132 dwords -> 16 banks

__global__ __launch_bounds__(128) void vq_argmin(
    const __bf16* __restrict__ xb, const __bf16* __restrict__ eb,
    const float* __restrict__ enorm, int* __restrict__ idx_out,
    float* __restrict__ idx_f_out) {
  __shared__ __align__(16) __bf16 lds[2][NTILE * LDS_STRIDE];  // 33 KB total

  const int t = threadIdx.x;                       // 0..127
  const int wave = t >> 5, lane = t & 31;
  const int l16 = lane & 15, half = lane >> 4;
  const int mrow = blockIdx.x * MTILE + wave * 16 + l16;

  // Resident A panel: this wave's 16 rows, full K=256 (8 fragments, 64 VGPRs)
  const __bf16* xrow = xb + (size_t)mrow * EMB_DIM;
  v16bf A[8];
#pragma unroll
  for (int f = 0; f < 8; ++f) A[f] = load_frag(xrow, 32 * f, half);

  float minv[8];
  int mini[8];
#pragma unroll
  for (int i = 0; i < 8; ++i) { minv[i] = 3.4e38f; mini[i] = 0; }

  // Async stage of one 32-row tile (16 KB = 1024 uint4) into LDS buffer b.
  // 128 threads x 8 async b128; padded rows of 33 uint4 (528 B).
  const unsigned lds_base = (unsigned)(uintptr_t)&lds[0][0];
  auto stage = [&](int nt, int b) {
    const char* src = (const char*)(eb + (size_t)nt * EMB_DIM);
    unsigned dbase = lds_base + (unsigned)b * (NTILE * LDS_STRIDE * 2);
#pragma unroll
    for (int i = 0; i < 8; ++i) {
      int li = t + i * 128;                        // uint4 index 0..1023
      int r = li >> 5, c = li & 31;
      async_copy_b128(dbase + (unsigned)(r * 33 + c) * 16, src + li * 16);
    }
  };

  const int NT = N_EMB / NTILE;                    // 256 tiles
  stage(0, 0);
  for (int ti = 0; ti < NT; ++ti) {
    int b = ti & 1;
    if (ti + 1 < NT) { stage(ti + 1, b ^ 1); wait_async_le8(); }
    else             { wait_async_0(); }
    __syncthreads();                               // tile ti visible to all

#pragma unroll
    for (int ns = 0; ns < NTILE / 16; ++ns) {
      const __bf16* ecol = &lds[b][(ns * 16 + l16) * LDS_STRIDE];
      v8f c = {};
#pragma unroll
      for (int f = 0; f < 8; ++f) {
        v16bf B = load_frag(ecol, 32 * f, half);
        c = __builtin_amdgcn_wmma_f32_16x16x32_bf16(
            false, A[f], false, B, (short)0, c, false, false);
      }
      int n = ti * NTILE + ns * 16 + l16;          // column owned by this lane
      float en = enorm[n];
#pragma unroll
      for (int i = 0; i < 8; ++i) {                // VGPR i -> row i / i+8
        float s = en - 2.0f * c[i];
        if (s < minv[i]) { minv[i] = s; mini[i] = n; }
      }
    }
    __syncthreads();                               // buf consumed -> restage ok
  }

  // Reduce (min,idx) across the 16-lane half holding each output row.
#pragma unroll
  for (int i = 0; i < 8; ++i) {
    float v = minv[i];
    int id = mini[i];
#pragma unroll
    for (int m = 8; m >= 1; m >>= 1) {
      float ov = __shfl_xor(v, m, 16);
      int oid = __shfl_xor(id, m, 16);
      if (ov < v || (ov == v && oid < id)) { v = ov; id = oid; }  // first-min
    }
    if (l16 == 0) {
      int row = blockIdx.x * MTILE + wave * 16 + half * 8 + i;
      idx_out[row] = id;
      idx_f_out[row] = (float)id;
    }
  }
}

// -------- gather quantized rows (f32 codebook) + loss partial sums --------
__global__ __launch_bounds__(256) void vq_gather_loss(
    const float* __restrict__ x, const float* __restrict__ cb,
    const int* __restrict__ idx, float* __restrict__ qout,
    float* __restrict__ acc) {
  int wave = threadIdx.x >> 5, lane = threadIdx.x & 31;
  int row = blockIdx.x * 8 + wave;                 // one wave per batch row
  int id = idx[row];
  const float4* q4 = (const float4*)(cb + (size_t)id * EMB_DIM);
  const float4* x4 = (const float4*)(x + (size_t)row * EMB_DIM);
  float4* o4 = (float4*)(qout + (size_t)row * EMB_DIM);
  float s = 0.0f;
#pragma unroll
  for (int i = 0; i < 2; ++i) {
    int c = lane + i * 32;
    float4 q = q4[c], xv = x4[c];
    o4[c] = q;                                     // quantized_st == quantized
    float dx = q.x - xv.x, dy = q.y - xv.y, dz = q.z - xv.z, dw = q.w - xv.w;
    s += dx * dx + dy * dy + dz * dz + dw * dw;
  }
#pragma unroll
  for (int m = 16; m >= 1; m >>= 1) s += __shfl_xor(s, m, 32);
  if (lane == 0) atomicAdd(acc, s);
}

__global__ void vq_finalize(const float* __restrict__ acc,
                            float* __restrict__ loss_out) {
  // vq + COMMIT*commit, both terms numerically equal -> 1.25 * mean
  *loss_out = 1.25f * (*acc) / (float)((size_t)BATCH * EMB_DIM);
}

extern "C" void kernel_launch(void* const* d_in, const int* in_sizes, int n_in,
                              void* d_out, int out_size, void* d_ws,
                              size_t ws_size, hipStream_t stream) {
  const float* x  = (const float*)d_in[0];         // [32768, 256] f32
  const float* cb = (const float*)d_in[1];         // [8192, 256]  f32

  // d_out: quantized_st [32768*256] | vq_loss [1] | indices [32768] (as f32)
  float* q_out    = (float*)d_out;
  float* loss_out = q_out + (size_t)BATCH * EMB_DIM;
  float* idxf_out = loss_out + 1;

  // Workspace layout (all 256B-aligned offsets)
  char* ws = (char*)d_ws;
  __bf16* xb    = (__bf16*)(ws);                         // 16 MB
  __bf16* eb    = (__bf16*)(ws + 16777216);              //  4 MB
  float*  enorm = (float*)(ws + 20971520);               // 32 KB
  int*    idx   = (int*)(ws + 21004288);                 // 128 KB
  float*  acc   = (float*)(ws + 21135360);               // 4 B

  vq_prep_x<<<(BATCH * EMB_DIM / 4 + 255) / 256, 256, 0, stream>>>(x, xb);
  vq_prep_e<<<N_EMB / 8, 256, 0, stream>>>(cb, eb, enorm, acc);
  vq_argmin<<<BATCH / MTILE, 128, 0, stream>>>(xb, eb, enorm, idx, idxf_out);
  vq_gather_loss<<<BATCH / 8, 256, 0, stream>>>(x, cb, idx, q_out, acc);
  vq_finalize<<<1, 1, 0, stream>>>(acc, loss_out);
}
